// VectorQuantizer_87540023427944
// MI455X (gfx1250) — matrix operand order repaired and checked by reading, not compiled, not run on hardware
//
#include <hip/hip_runtime.h>
#include <hip/hip_bf16.h>
#include <math.h>

// ---------------------------------------------------------------------------
// Vector quantizer for MI455X (gfx1250, wave32, WMMA bf16 16x16x32 + TDM)
//   B=16 S=2048 L=E=512 N=2048, BS=32768
// ---------------------------------------------------------------------------

#define BS   32768
#define EDIM 512
#define NCB  2048
#define LDIM 512

typedef __attribute__((ext_vector_type(16))) __bf16 v16bf;
typedef __attribute__((ext_vector_type(8)))  __bf16 v8bf;
typedef __attribute__((ext_vector_type(8)))  float  v8f;
typedef __attribute__((ext_vector_type(4)))  unsigned int v4u;
typedef __attribute__((ext_vector_type(8)))  int v8i;
typedef __attribute__((ext_vector_type(4)))  int v4i;

#if defined(__has_builtin)
#if __has_builtin(__builtin_amdgcn_tensor_load_to_lds) && \
    __has_builtin(__builtin_amdgcn_s_wait_tensorcnt)
#define USE_TDM 1
#endif
#endif
#ifndef USE_TDM
#define USE_TDM 0
#endif

// ---- fragment loaders (contiguous k = 16h + s in A and B, wave32) ------------------
// A (16x32, MxK): lanes 0-15 row M=lane hold K 0..15; lanes 16-31 row
// M=lane-16 hold K 16..31.
__device__ __forceinline__ v16bf load_a_bf16(const __bf16* base, int ld,
                                             int m0, int k0, int lane) {
  const int half = lane >> 4, mr = lane & 15;
  const __bf16* p = base + (size_t)(m0 + mr) * ld;
  v8bf lo = *(const v8bf*)(p + k0 + half * 16);
  v8bf hi = *(const v8bf*)(p + k0 + 8 + half * 16);
  v16bf f;
#pragma unroll
  for (int i = 0; i < 8; ++i) { f[i] = lo[i]; f[i + 8] = hi[i]; }
  return f;
}

// A fragment sourced from fp32 (on-the-fly bf16 convert)
__device__ __forceinline__ v16bf load_a_f32(const float* base, int ld,
                                            int m0, int k0, int lane) {
  const int half = lane >> 4, mr = lane & 15;
  const float* p = base + (size_t)(m0 + mr) * ld;
  v16bf f;
#pragma unroll
  for (int i = 0; i < 8; ++i) {
    f[i]     = (__bf16)p[k0 + half * 16 + i];
    f[i + 8] = (__bf16)p[k0 + 8 + half * 16 + i];
  }
  return f;
}

// B (32x16, KxN) with column n stored contiguously (row n of the source
// matrix): lanes 0-15 col N=lane hold K 0..15; lanes 16-31 hold K 16..31.
__device__ __forceinline__ v16bf load_b_bf16(const __bf16* bmat, int ld,
                                             int n0, int k0, int lane) {
  const int half = lane >> 4, nc = lane & 15;
  return *(const v16bf*)(bmat + (size_t)(n0 + nc) * ld + k0 + half * 16);
}

__device__ __forceinline__ v8f wmma_bf16(v16bf a, v16bf b, v8f c) {
  return __builtin_amdgcn_wmma_f32_16x16x32_bf16(false, a, false, b,
                                                 (short)0, c, false, false);
}

#if USE_TDM
// ---- Tensor Data Mover: 2D tile (16-bit elements) global -> LDS ------------
// D# per ISA ch.8: group0 {count, lds_addr, global_addr(57b), type=2},
// group1 {wg_mask=0, data_size=2B, tensor_dim0/1, tile_dim0/1, dim0 stride}.
__device__ __forceinline__ void tdm_load_2d(unsigned lds_addr, const void* gp,
                                            unsigned tile_d0, unsigned tile_d1,
                                            unsigned tensor_d0, unsigned tensor_d1,
                                            unsigned stride0) {
  unsigned long long ga = (unsigned long long)(uintptr_t)gp;
  v4u g0;
  g0[0] = 1u;                                   // count=1 valid descriptor
  g0[1] = lds_addr;                             // LDS byte address
  g0[2] = (unsigned)ga;                         // global addr [31:0]
  g0[3] = (unsigned)((ga >> 32) & 0x1FFFFFFu) | (2u << 30);  // [56:32] | type=2
  v8i g1;
  g1[0] = (int)(1u << 16);                      // data_size=1 -> 2 bytes
  g1[1] = (int)((tensor_d0 & 0xFFFFu) << 16);   // tensor_dim0[15:0]
  g1[2] = (int)(((tensor_d0 >> 16) & 0xFFFFu) | ((tensor_d1 & 0xFFFFu) << 16));
  g1[3] = (int)(((tensor_d1 >> 16) & 0xFFFFu) | ((tile_d0 & 0xFFFFu) << 16));
  g1[4] = (int)(tile_d1 & 0xFFFFu);             // tile_dim1, tile_dim2=0
  g1[5] = (int)stride0;                         // tensor_dim0_stride[31:0]
  g1[6] = 0;                                    // stride0 hi, stride1 lo
  g1[7] = 0;
  v4i gz = {0, 0, 0, 0};
#if defined(__clang_major__) && (__clang_major__ >= 23)
  v8i gz8 = {0, 0, 0, 0, 0, 0, 0, 0};
  __builtin_amdgcn_tensor_load_to_lds(g0, g1, gz, gz, gz8, 0);
#else
  __builtin_amdgcn_tensor_load_to_lds(g0, g1, gz, gz, 0);
#endif
}
#endif  // USE_TDM

// ---- prep kernels ----------------------------------------------------------
__global__ void k_cvt_bf16(const float* __restrict__ src,
                           unsigned short* __restrict__ dst_raw, int n) {
  __bf16* dst = (__bf16*)dst_raw;
  for (int i = blockIdx.x * blockDim.x + threadIdx.x; i < n;
       i += gridDim.x * blockDim.x)
    dst[i] = (__bf16)src[i];
}

// normalize one codebook row per block -> bf16
__global__ void k_emb_norm(const float* __restrict__ emb,
                           unsigned short* __restrict__ emb_bf_raw) {
  __bf16* emb_bf = (__bf16*)emb_bf_raw;
  const int row = blockIdx.x;
  const float* p = emb + (size_t)row * EDIM;
  __shared__ float ssum;
  if (threadIdx.x == 0) ssum = 0.f;
  __syncthreads();
  float s = 0.f;
  for (int i = threadIdx.x; i < EDIM; i += blockDim.x) { float v = p[i]; s += v * v; }
#pragma unroll
  for (int off = 16; off > 0; off >>= 1) s += __shfl_down(s, off, 32);
  if ((threadIdx.x & 31) == 0) atomicAdd(&ssum, s);
  __syncthreads();
  const float inv = 1.0f / fmaxf(sqrtf(ssum), 1e-12f);
  for (int i = threadIdx.x; i < EDIM; i += blockDim.x)
    emb_bf[(size_t)row * EDIM + i] = (__bf16)(p[i] * inv);
}

// ---- GEMM1 fused with row L2-normalization ---------------------------------
// block: 16 rows x full E=512 (8 waves x 4 n-tiles); grid: BS/16
__global__ void k_inproj_norm(const float* __restrict__ z,
                              const unsigned short* __restrict__ Wbf_raw,
                              const float* __restrict__ b_in,
                              unsigned short* __restrict__ zf_raw) {
  const __bf16* Wbf = (const __bf16*)Wbf_raw;
  __bf16* zf = (__bf16*)zf_raw;
  const int lane = threadIdx.x & 31, wave = threadIdx.x >> 5;
  const int m0 = blockIdx.x * 16;
  const int half = lane >> 4, nc = lane & 15;
  __shared__ float snorm[16];
  if (threadIdx.x < 16) snorm[threadIdx.x] = 0.f;
  __syncthreads();

  v8f acc[4] = {v8f{}, v8f{}, v8f{}, v8f{}};
  for (int k0 = 0; k0 < EDIM; k0 += 32) {
    __builtin_prefetch(z + (size_t)(m0 + (lane & 15)) * EDIM + k0 + 64, 0, 3);
    v16bf a = load_a_f32(z, EDIM, m0, k0, lane);  // shared by 4 n-tiles
#pragma unroll
    for (int t = 0; t < 4; ++t)
      acc[t] = wmma_bf16(a, load_b_bf16(Wbf, EDIM, wave * 64 + t * 16, k0, lane), acc[t]);
  }

  float vals[4][8];
#pragma unroll
  for (int t = 0; t < 4; ++t) {
    const float bias = b_in[wave * 64 + t * 16 + nc];
#pragma unroll
    for (int r = 0; r < 8; ++r) vals[t][r] = acc[t][r] + bias;
  }
#pragma unroll
  for (int r = 0; r < 8; ++r) {   // per-lane partial sum of squares per row
    float s = 0.f;
#pragma unroll
    for (int t = 0; t < 4; ++t) s += vals[t][r] * vals[t][r];
    atomicAdd(&snorm[r + 8 * half], s);   // ds_add_f32
  }
  __syncthreads();
#pragma unroll
  for (int r = 0; r < 8; ++r) {
    const int row = r + 8 * half;
    const float inv = 1.0f / fmaxf(sqrtf(snorm[row]), 1e-12f);
    const size_t base = (size_t)(m0 + row) * EDIM;
#pragma unroll
    for (int t = 0; t < 4; ++t)
      zf[base + wave * 64 + t * 16 + nc] = (__bf16)(vals[t][r] * inv);
  }
}

// ---- GEMM2: cosine similarity vs all 2048 codes, fused argmax --------------
// block: 16 rows x full N=2048. K is the outer loop; the codebook K-slice
// (2048x32 bf16 = 128 KB) is DMA'd into LDS by the TDM, double-buffered, and
// the 16x512 A panel (16 KB) is staged once. Each wave owns 16 n-tiles
// (128 accumulator VGPRs) and reuses one A fragment across all 16 WMMAs.
#define EBUF_ELEMS (NCB * 32)                       // 64 K elements, 128 KB
#define DIST_LDS_BYTES (16 * EDIM * 2 + 2 * EBUF_ELEMS * 2)  // 16KB + 256KB

__device__ __forceinline__ unsigned float_key(float v) {
  unsigned b = __float_as_uint(v);
  return (b & 0x80000000u) ? ~b : (b | 0x80000000u);  // monotone total order
}

__global__ void k_dist_argmax(const unsigned short* __restrict__ zf_raw,
                              const unsigned short* __restrict__ emb_raw,
                              int* __restrict__ idx_out,
                              float* __restrict__ idx_out_f) {
  const __bf16* zf = (const __bf16*)zf_raw;
  const __bf16* emb_bf = (const __bf16*)emb_raw;
  const int tid = threadIdx.x;
  const int lane = tid & 31, wave = tid >> 5;
  const int m0 = blockIdx.x * 16;
  const int half = lane >> 4, nc = lane & 15;

  extern __shared__ char dyn_smem[];
  __bf16* abuf = (__bf16*)dyn_smem;                        // 16 x 512
  __bf16* ebase = (__bf16*)(dyn_smem + 16 * EDIM * 2);     // 2 x 128 KB
  __shared__ unsigned long long best[16];
  if (tid < 16) best[tid] = 0ull;  // key(-inf) ~ 0

#if USE_TDM
  if (tid < 32) {  // wave 0 issues DMA; wave-uniform branch
    // A panel: 16 contiguous rows = flat 1D copy of 8192 elements
    tdm_load_2d((unsigned)(uintptr_t)abuf, zf + (size_t)m0 * EDIM,
                /*tile*/ 16 * EDIM, 0, /*tensor*/ 16 * EDIM, 1, 16 * EDIM);
    // codebook K-slice chunk 0: 2048 rows x 32 cols, row stride 512
    tdm_load_2d((unsigned)(uintptr_t)ebase, emb_bf,
                /*tile*/ 32, NCB, /*tensor*/ EDIM, NCB, EDIM);
  }
#else
  for (int i = tid; i < 16 * EDIM / 8; i += 256)  // cooperative A-panel copy
    ((uint4*)abuf)[i] = ((const uint4*)(zf + (size_t)m0 * EDIM))[i];
#endif

  v8f acc[16];
#pragma unroll
  for (int t = 0; t < 16; ++t) acc[t] = v8f{};

  for (int c = 0; c < 16; ++c) {        // K chunks of 32
    __bf16* cur = ebase + (size_t)(c & 1) * EBUF_ELEMS;
#if USE_TDM
    if (tid < 32) {
      if (c + 1 < 16) {
        tdm_load_2d((unsigned)(uintptr_t)(ebase + (size_t)((c + 1) & 1) * EBUF_ELEMS),
                    emb_bf + (c + 1) * 32, 32, NCB, EDIM, NCB, EDIM);
        __builtin_amdgcn_s_wait_tensorcnt(1);  // chunk c (and A panel) done
      } else {
        __builtin_amdgcn_s_wait_tensorcnt(0);
      }
    }
    __syncthreads();
#else
    __syncthreads();  // previous chunk's consumers done before overwrite
    for (int i = tid; i < NCB * 4; i += 256) {
      const int n = i >> 2, q = i & 3;
      ((uint4*)cur)[i] =
          *(const uint4*)(emb_bf + (size_t)n * EDIM + c * 32 + q * 8);
    }
    __syncthreads();
#endif
    // one A fragment feeds all 16 n-tiles of this wave
    v16bf a = load_a_bf16(abuf, EDIM, 0, c * 32, lane);
#pragma unroll
    for (int t = 0; t < 16; ++t) {
      const int n = wave * 256 + t * 16 + nc;
      v16bf b = *(const v16bf*)(cur + (size_t)n * 32 + half * 16);
      acc[t] = wmma_bf16(a, b, acc[t]);
    }
    __syncthreads();  // all waves done with `cur` before it is re-filled
  }

  // fused argmax: max dot == min distance; ~n in low bits -> smallest index
  // wins ties (matches argmin first-occurrence).
  unsigned long long lbest[8];
#pragma unroll
  for (int r = 0; r < 8; ++r) lbest[r] = 0ull;
#pragma unroll
  for (int t = 0; t < 16; ++t) {
    const unsigned n = (unsigned)(wave * 256 + t * 16 + nc);
#pragma unroll
    for (int r = 0; r < 8; ++r) {
      unsigned long long p =
          ((unsigned long long)float_key(acc[t][r]) << 32) | (unsigned)(~n);
      if (p > lbest[r]) lbest[r] = p;
    }
  }
#pragma unroll
  for (int r = 0; r < 8; ++r)
    atomicMax(&best[r + 8 * half], lbest[r]);  // ds_max_u64
  __syncthreads();
  if (tid < 16) {
    const unsigned n = ~(unsigned)(best[tid] & 0xffffffffull);
    idx_out[m0 + tid] = (int)n;
    idx_out_f[m0 + tid] = (float)n;  // second tuple output
  }
}

// ---- GEMM3: gather codebook rows, output projection ------------------------
__global__ void k_outproj(const int* __restrict__ idx,
                          const unsigned short* __restrict__ emb_raw,
                          const unsigned short* __restrict__ Wout_raw,
                          const float* __restrict__ b_out,
                          float* __restrict__ out) {
  const __bf16* emb_bf = (const __bf16*)emb_raw;
  const __bf16* Wbf = (const __bf16*)Wout_raw;
  const int lane = threadIdx.x & 31, wave = threadIdx.x >> 5;
  const int m0 = blockIdx.x * 16;
  const int half = lane >> 4, nc = lane & 15, mr = lane & 15;
  // each lane's A rows come from the gathered codebook entry for row m0+mr
  const __bf16* arow = emb_bf + (size_t)idx[m0 + mr] * EDIM;

  v8f acc[4] = {v8f{}, v8f{}, v8f{}, v8f{}};
  for (int k0 = 0; k0 < EDIM; k0 += 32) {
    v16bf a;
    v8bf lo = *(const v8bf*)(arow + k0 + half * 16);
    v8bf hi = *(const v8bf*)(arow + k0 + 8 + half * 16);
#pragma unroll
    for (int i = 0; i < 8; ++i) { a[i] = lo[i]; a[i + 8] = hi[i]; }
#pragma unroll
    for (int t = 0; t < 4; ++t)
      acc[t] = wmma_bf16(a, load_b_bf16(Wbf, EDIM, wave * 64 + t * 16, k0, lane), acc[t]);
  }
#pragma unroll
  for (int r = 0; r < 8; ++r) {
    const int row = r + 8 * half;
    const size_t base = (size_t)(m0 + row) * LDIM;
#pragma unroll
    for (int t = 0; t < 4; ++t) {
      const int n = wave * 64 + t * 16 + nc;
      out[base + n] = acc[t][r] + b_out[n];  // STE output == zq
    }
  }
}

// ---------------------------------------------------------------------------
extern "C" void kernel_launch(void* const* d_in, const int* in_sizes, int n_in,
                              void* d_out, int out_size, void* d_ws, size_t ws_size,
                              hipStream_t stream) {
  const float* z     = (const float*)d_in[0];
  // d_in[1] = mask (unused by forward body)
  const float* W_in  = (const float*)d_in[2];
  const float* b_in  = (const float*)d_in[3];
  const float* W_out = (const float*)d_in[4];
  const float* b_out = (const float*)d_in[5];
  const float* emb   = (const float*)d_in[6];
  float* out = (float*)d_out;

  char* ws = (char*)d_ws;
  unsigned short* W_in_bf  = (unsigned short*)(ws);                 // 512 KB
  unsigned short* W_out_bf = (unsigned short*)(ws + (512u << 10));  // 512 KB
  unsigned short* emb_bf   = (unsigned short*)(ws + (1024u << 10)); // 2 MB
  unsigned short* zf_bf    = (unsigned short*)(ws + (3072u << 10)); // 32 MB
  int*            idx_ws   = (int*)(ws + (3072u << 10) + (size_t)BS * EDIM * 2);

  k_cvt_bf16<<<512, 256, 0, stream>>>(W_in, W_in_bf, EDIM * LDIM);
  k_cvt_bf16<<<512, 256, 0, stream>>>(W_out, W_out_bf, LDIM * EDIM);
  k_emb_norm<<<NCB, 256, 0, stream>>>(emb, emb_bf);
  k_inproj_norm<<<BS / 16, 256, 0, stream>>>(z, W_in_bf, b_in, zf_bf);
  k_dist_argmax<<<BS / 16, 256, DIST_LDS_BYTES, stream>>>(
      zf_bf, emb_bf, idx_ws, out + (size_t)BS * LDIM);
  k_outproj<<<BS / 16, 256, 0, stream>>>(idx_ws, emb_bf, W_out_bf, b_out, out);
}
